// HymbaSparseMoeBlock_40561671144016
// MI455X (gfx1250) — matrix operand order repaired and checked by reading, not compiled, not run on hardware
//
#include <hip/hip_runtime.h>
#include <cstdint>
#include <cstddef>

// ---------------------------------------------------------------------------
// MoE (top-2 of 8 experts), D=1024, F=2048, T=4096 tokens.
// Sparse-routed grouped GEMMs in bf16 via v_wmma_f32_16x16x32_bf16 (gfx1250),
// weight tiles streamed into a 3-deep LDS ring by the Tensor Data Mover.
// Incremental TDM addressing keeps the scalar pipe off the issue critical path.
// ---------------------------------------------------------------------------

typedef __attribute__((ext_vector_type(16))) __bf16 v16bf;
typedef __attribute__((ext_vector_type(8)))  float  v8f;
typedef __attribute__((ext_vector_type(4)))  unsigned int u32x4;
typedef __attribute__((ext_vector_type(8)))  int  i32x8;
typedef __attribute__((ext_vector_type(4)))  int  i32x4;

static constexpr int T = 4096;   // B*S tokens
static constexpr int D = 1024;
static constexpr int F = 2048;
static constexpr int E = 8;

#if __has_builtin(__builtin_amdgcn_tensor_load_to_lds) && __has_builtin(__builtin_amdgcn_s_wait_tensorcnt)
#define USE_TDM 1
#else
#define USE_TDM 0
#endif

// ---- workspace layout (bytes) --------------------------------------------
static constexpr size_t XB_OFF    = 0;                               // x bf16 [T,D]
static constexpr size_t XB_SZ     = (size_t)T * D * 2;
static constexpr size_t GATE_OFF  = XB_OFF + XB_SZ;                  // [E,F,D] bf16
static constexpr size_t W_SZ      = (size_t)E * F * D * 2;
static constexpr size_t UP_OFF    = GATE_OFF + W_SZ;                 // [E,F,D] bf16
static constexpr size_t DOWN_OFF  = UP_OFF + W_SZ;                   // [E,D,F] bf16
static constexpr size_t WFULL_OFF = DOWN_OFF + W_SZ;                 // [T,E] f32 combine w
static constexpr size_t CNT_OFF   = WFULL_OFF + (size_t)T * E * 4;   // [E] int
static constexpr size_t OFFS_OFF  = CNT_OFF + 256;                   // [E+1] int
static constexpr size_t TOK_OFF   = OFFS_OFF + 256;                  // [E,T] int
static constexpr size_t HBUF_OFF  = TOK_OFF + (size_t)E * T * 4;     // [2T,F] bf16 h scratch
// total ~143 MB

#define DEV __device__ __forceinline__

DEV __bf16 f2bf(float f) {            // round-to-nearest-even fp32 -> bf16
  uint32_t u = __float_as_uint(f);
  uint32_t r = (u + 0x7FFFu + ((u >> 16) & 1u)) >> 16;
  uint16_t s = (uint16_t)r;
  __bf16 b;
  __builtin_memcpy(&b, &s, 2);
  return b;
}

union ABfrag { v16bf v; uint4 u[2]; };

DEV v8f wmma_bf16(v16bf a, v16bf b, v8f c) {
  return __builtin_amdgcn_wmma_f32_16x16x32_bf16(
      false, a, false, b, (short)0, c, false, false);
}

#if USE_TDM
DEV uint32_t lds_off(const void* p) {   // generic shared ptr -> LDS byte address
  return (uint32_t)(uintptr_t)p;
}

// One 2D tile: 128 rows x 32 bf16, row stride `stride_elems`.
// D# bitfields per CDNA5 ISA 8.3/8.4. 6-arg builtin (amdgpu-toolchain).
DEV void tdm_load_tile_128x32(uint32_t lds_addr, uint64_t ga,
                              uint32_t stride_elems) {
  u32x4 g0 = { 1u,                                   // count=1, user desc
               lds_addr,                             // lds_addr
               (uint32_t)ga,                         // global_addr[31:0]
               (uint32_t)((ga >> 32) & 0x01FFFFFFu) | 0x80000000u }; // [56:32] | type=2
  const uint32_t td0 = 1u << 20, td1 = 1u << 20;     // tensor dims: large (no OOB)
  i32x8 g1 = { (int)0x00010000u,                                     // data_size=1 (2B)
               (int)((td0 & 0xFFFFu) << 16),                         // tensor_dim0 lo
               (int)(((td0 >> 16) & 0xFFFFu) | ((td1 & 0xFFFFu) << 16)),
               (int)(((td1 >> 16) & 0xFFFFu) | (32u << 16)),         // tile_dim0=32
               (int)128u,                                            // tile_dim1=128
               (int)stride_elems,                                    // dim0 stride lo
               0, 0 };
  i32x4 z4 = { 0, 0, 0, 0 };
  i32x8 z8 = { 0, 0, 0, 0, 0, 0, 0, 0 };
  __builtin_amdgcn_tensor_load_to_lds(g0, g1, z4, z4, z8, 0);
}
#endif

// ---------------------------------------------------------------------------
__global__ __launch_bounds__(256) void cvt_bf16_kernel(
    const float* __restrict__ src, __bf16* __restrict__ dst, size_t n) {
  size_t i = ((size_t)blockIdx.x * 256 + threadIdx.x) * 4;
  if (i >= n) return;
  float4 v = *(const float4*)(src + i);
  __bf16 o[4] = { f2bf(v.x), f2bf(v.y), f2bf(v.z), f2bf(v.w) };
  uint2 p; __builtin_memcpy(&p, o, 8);
  *(uint2*)(dst + i) = p;
}

// ---------------------------------------------------------------------------
// Router: one wave32 per token. logits -> softmax -> top2 -> gather lists.
// ---------------------------------------------------------------------------
__global__ __launch_bounds__(256) void router_kernel(
    const float* __restrict__ x, const float* __restrict__ rw,
    float* __restrict__ logits_out, float* __restrict__ wfull,
    int* __restrict__ counts, int* __restrict__ tok_list) {
  int t = blockIdx.x * 8 + (threadIdx.x >> 5);
  int lane = threadIdx.x & 31;
  const float* xr = x + (size_t)t * D;

  float acc[E];
#pragma unroll
  for (int e = 0; e < E; e++) acc[e] = 0.f;

  for (int i = 0; i < D / 32; i++) {
    float xv = xr[lane + i * 32];
#pragma unroll
    for (int e = 0; e < E; e++) acc[e] += xv * rw[e * D + lane + i * 32];
  }
#pragma unroll
  for (int e = 0; e < E; e++) {
#pragma unroll
    for (int off = 16; off > 0; off >>= 1)
      acc[e] += __shfl_xor(acc[e], off, 32);
  }

  if (lane == 0) {
    float mx = acc[0];
#pragma unroll
    for (int e = 1; e < E; e++) mx = fmaxf(mx, acc[e]);
    float p[E], s = 0.f;
#pragma unroll
    for (int e = 0; e < E; e++) { p[e] = __expf(acc[e] - mx); s += p[e]; }
    float inv = 1.f / s;

    int i1 = 0;
#pragma unroll
    for (int e = 1; e < E; e++) if (p[e] > p[i1]) i1 = e;
    int i2 = (i1 == 0) ? 1 : 0;
#pragma unroll
    for (int e = 0; e < E; e++) if (e != i1 && p[e] > p[i2]) i2 = e;

#pragma unroll
    for (int e = 0; e < E; e++) {
      logits_out[(size_t)t * E + e] = acc[e];
      wfull[(size_t)t * E + e] = (e == i1 || e == i2) ? p[e] * inv : 0.f;
    }
    int p1 = atomicAdd(&counts[i1], 1); tok_list[i1 * T + p1] = t;
    int p2 = atomicAdd(&counts[i2], 1); tok_list[i2 * T + p2] = t;
  }
}

__global__ void scan_kernel(const int* __restrict__ counts, int* __restrict__ offs) {
  if (threadIdx.x == 0 && blockIdx.x == 0) {
    int s = 0;
    for (int e = 0; e < E; e++) { offs[e] = s; s += counts[e]; }
    offs[E] = s;
  }
}

// ---------------------------------------------------------------------------
// FFN1: per (expert, 64-token tile): g/u = X @ Wg^T / Wu^T, h = silu(g)*u.
// Block tile 64x128/chunk. A frags from global; W tiles via TDM 3-ring.
// ---------------------------------------------------------------------------
__global__ __launch_bounds__(256) void ffn1_kernel(
    const __bf16* __restrict__ xb, const __bf16* __restrict__ gate_b,
    const __bf16* __restrict__ up_b, const int* __restrict__ tok_list,
    const int* __restrict__ counts, const int* __restrict__ offs,
    __bf16* __restrict__ h_buf) {
  int e = blockIdx.y;
  int cnt = counts[e];
  int base = blockIdx.x * 64;
  if (base >= cnt) return;
  int nrow = cnt - base; if (nrow > 64) nrow = 64;
  int slotbase = offs[e] + base;

  constexpr int SLOT = 128 * 32;                 // elements per ring slot
  __shared__ int    toks[64];
  __shared__ __bf16 WgR[3 * SLOT];
  __shared__ __bf16 WuR[3 * SLOT];

  int tid = threadIdx.x;
  if (tid < 64) {
    int idx = base + tid; if (idx > cnt - 1) idx = cnt - 1;
    toks[tid] = tok_list[e * T + idx];
  }
  __syncthreads();

  int lane = tid & 31, wave = tid >> 5;
  int wm = wave & 1, wn = wave >> 1;            // 2 (M) x 4 (N) waves
  int am = lane & 15, ak = (lane >> 4) * 16;
  int bn = lane & 15, bk = (lane >> 4) * 16;
  int mhi = (lane & 16) ? 8 : 0;

  const __bf16* arow0 = xb + (size_t)toks[wm * 32 + am] * D;
  const __bf16* arow1 = xb + (size_t)toks[wm * 32 + 16 + am] * D;
  const __bf16* wbase = gate_b + (size_t)e * F * D;
  const __bf16* ubase = up_b   + (size_t)e * F * D;

  constexpr int STEPS = (F / 128) * (D / 32);    // 512
  constexpr uint64_t JUMP = ((uint64_t)128 * D - 31 * 32) * 2;  // block-cross delta

  uint32_t coff = 0;                 // consumer ring offset (elements)
  uint32_t noff = SLOT;              // producer target ring offset
#if USE_TDM
  uint32_t wgL = lds_off(&WgR[0]);
  uint32_t wuL = lds_off(&WuR[0]);
  uint64_t gaG = (uint64_t)(uintptr_t)wbase;     // addr of next issue (step 1 after +64)
  uint64_t gaU = (uint64_t)(uintptr_t)ubase;
  if (tid < 32) {                                // prologue: step 0 into slot 0
    tdm_load_tile_128x32(wgL, gaG, D);
    tdm_load_tile_128x32(wuL, gaU, D);
  }
  gaG += 64; gaU += 64;                          // -> address of step 1 tile
#endif

  const v8f vzero = {0.f, 0.f, 0.f, 0.f, 0.f, 0.f, 0.f, 0.f};
  int s = 0;
  for (int fbase = 0; fbase < F; fbase += 128) {
    v8f gacc[2][2], uacc[2][2];
#pragma unroll
    for (int im = 0; im < 2; im++)
#pragma unroll
      for (int in = 0; in < 2; in++) { gacc[im][in] = vzero; uacc[im][in] = vzero; }

    for (int kk = 0; kk < D; kk += 32, s++) {
#if USE_TDM
      if (tid < 32) {
        if (s + 1 < STEPS) {
          tdm_load_tile_128x32(wgL + noff * 2, gaG, D);
          tdm_load_tile_128x32(wuL + noff * 2, gaU, D);
          __builtin_amdgcn_s_wait_tensorcnt((short)2);  // current pair done
        } else {
          __builtin_amdgcn_s_wait_tensorcnt((short)0);
        }
      }
      {  // uniform incremental update for the *next* issue (step s+2)
        uint64_t d = (((s + 2) & 31) == 0) ? JUMP : 64;
        gaG += d; gaU += d;
      }
#else
      {
        int wr = tid >> 1, wc = (tid & 1) * 16;
        size_t wrow = ((size_t)fbase + wr) * D + kk;
        *(uint4*)&WgR[coff + wr * 32 + wc]     = *(const uint4*)(wbase + wrow + wc);
        *(uint4*)&WgR[coff + wr * 32 + wc + 8] = *(const uint4*)(wbase + wrow + wc + 8);
        *(uint4*)&WuR[coff + wr * 32 + wc]     = *(const uint4*)(ubase + wrow + wc);
        *(uint4*)&WuR[coff + wr * 32 + wc + 8] = *(const uint4*)(ubase + wrow + wc + 8);
        if (kk + 32 < D) {
          __builtin_prefetch(wbase + wrow + 32 + wc, 0, 3);
          __builtin_prefetch(ubase + wrow + 32 + wc, 0, 3);
        }
      }
#endif
      __syncthreads();   // ring depth 3 + issue-ahead 1 -> one barrier/step safe

      ABfrag a[2];
      a[0].u[0] = *(const uint4*)(arow0 + kk + ak);
      a[0].u[1] = *(const uint4*)(arow0 + kk + ak + 8);
      a[1].u[0] = *(const uint4*)(arow1 + kk + ak);
      a[1].u[1] = *(const uint4*)(arow1 + kk + ak + 8);

#pragma unroll
      for (int in = 0; in < 2; in++) {
        ABfrag bg, bu;
        const __bf16* bgr = &WgR[coff + (wn * 32 + in * 16 + bn) * 32 + bk];
        bg.u[0] = *(const uint4*)(bgr);
        bg.u[1] = *(const uint4*)(bgr + 8);
        const __bf16* bur = &WuR[coff + (wn * 32 + in * 16 + bn) * 32 + bk];
        bu.u[0] = *(const uint4*)(bur);
        bu.u[1] = *(const uint4*)(bur + 8);
#pragma unroll
        for (int im = 0; im < 2; im++) {
          gacc[im][in] = wmma_bf16(a[im].v, bg.v, gacc[im][in]);
          uacc[im][in] = wmma_bf16(a[im].v, bu.v, uacc[im][in]);
        }
      }

      coff = noff;                                   // rotate ring
      noff = (noff == 2 * SLOT) ? 0u : noff + SLOT;
    }

    // SwiGLU epilogue -> bf16 h scratch
#pragma unroll
    for (int im = 0; im < 2; im++)
#pragma unroll
      for (int in = 0; in < 2; in++)
#pragma unroll
        for (int j = 0; j < 8; j++) {
          int m = wm * 32 + im * 16 + j + mhi;
          if (m < nrow) {
            float g = gacc[im][in][j];
            float u = uacc[im][in][j];
            float h = (g / (1.f + __expf(-g))) * u;
            int n = wn * 32 + in * 16 + (lane & 15);
            h_buf[(size_t)(slotbase + m) * F + fbase + n] = f2bf(h);
          }
        }
  }
}

// ---------------------------------------------------------------------------
// FFN2: y = h @ down^T (down [E,D,F], K=F contiguous). Block N-chunk = 256
// (2 TDM tiles / step, 8 WMMAs per barrier). Deterministic atomic combine.
// ---------------------------------------------------------------------------
__global__ __launch_bounds__(256) void ffn2_kernel(
    const __bf16* __restrict__ h_buf, const __bf16* __restrict__ down_b,
    const int* __restrict__ tok_list, const int* __restrict__ counts,
    const int* __restrict__ offs, const float* __restrict__ wfull,
    float* __restrict__ out) {
  int e = blockIdx.y;
  int cnt = counts[e];
  int base = blockIdx.x * 64;
  if (base >= cnt) return;
  int nrow = cnt - base; if (nrow > 64) nrow = 64;
  int slotbase = offs[e] + base;

  constexpr int TILE = 128 * 32;                 // one 128x32 tile (elements)
  constexpr int SLOT = 2 * TILE;                 // ring slot = 2 tiles (256 cols)
  __shared__ int    toks[64];
  __shared__ float  wrow[64];
  __shared__ __bf16 WdR[3 * SLOT];               // 48 KB

  int tid = threadIdx.x;
  if (tid < 64) {
    int idx = base + tid; if (idx > cnt - 1) idx = cnt - 1;
    int tk = tok_list[e * T + idx];
    toks[tid] = tk;
    wrow[tid] = wfull[(size_t)tk * E + e];
  }
  __syncthreads();

  int lane = tid & 31, wave = tid >> 5;
  int wm = wave & 1, wn = wave >> 1;             // 2 (M) x 4 (N=64 each) waves
  int am = lane & 15, ak = (lane >> 4) * 16;
  int bn = lane & 15, bk = (lane >> 4) * 16;
  int mhi = (lane & 16) ? 8 : 0;

  int r0 = wm * 32 + am;      if (r0 > nrow - 1) r0 = nrow - 1;
  int r1 = wm * 32 + 16 + am; if (r1 > nrow - 1) r1 = nrow - 1;
  const __bf16* hrow0 = h_buf + (size_t)(slotbase + r0) * F;
  const __bf16* hrow1 = h_buf + (size_t)(slotbase + r1) * F;
  const __bf16* dbase0 = down_b + (size_t)e * D * F;

  constexpr int STEPS = (D / 256) * (F / 32);    // 256
  constexpr uint64_t TILE2B = (uint64_t)128 * F * 2;              // 2nd tile offset
  constexpr uint64_t JUMP = ((uint64_t)256 * F - 63 * 32) * 2;    // chunk-cross delta

  uint32_t coff = 0, noff = SLOT;
#if USE_TDM
  uint32_t wdL = lds_off(&WdR[0]);
  uint64_t gaD = (uint64_t)(uintptr_t)dbase0;
  if (tid < 32) {
    tdm_load_tile_128x32(wdL, gaD, F);
    tdm_load_tile_128x32(wdL + TILE * 2, gaD + TILE2B, F);
  }
  gaD += 64;
#endif

  const v8f vzero = {0.f, 0.f, 0.f, 0.f, 0.f, 0.f, 0.f, 0.f};
  int s = 0;
  for (int dbase = 0; dbase < D; dbase += 256) {
    v8f yacc[2][4];
#pragma unroll
    for (int im = 0; im < 2; im++)
#pragma unroll
      for (int in = 0; in < 4; in++) yacc[im][in] = vzero;

    for (int kk = 0; kk < F; kk += 32, s++) {
#if USE_TDM
      if (tid < 32) {
        if (s + 1 < STEPS) {
          tdm_load_tile_128x32(wdL + noff * 2, gaD, F);
          tdm_load_tile_128x32(wdL + noff * 2 + TILE * 2, gaD + TILE2B, F);
          __builtin_amdgcn_s_wait_tensorcnt((short)2);
        } else {
          __builtin_amdgcn_s_wait_tensorcnt((short)0);
        }
      }
      gaD += (((s + 2) & 63) == 0) ? JUMP : 64;
#else
      {
        int t2 = tid >> 7, r = tid & 127;        // 256 rows, one row per thread
        size_t drow = ((size_t)dbase + t2 * 128 + r) * F + kk;
#pragma unroll
        for (int c = 0; c < 32; c += 8)
          *(uint4*)&WdR[coff + t2 * TILE + r * 32 + c] =
              *(const uint4*)(dbase0 + drow + c);
        if (kk + 32 < F)
          __builtin_prefetch(dbase0 + drow + 32, 0, 3);
      }
#endif
      __syncthreads();

      ABfrag a[2];
      a[0].u[0] = *(const uint4*)(hrow0 + kk + ak);
      a[0].u[1] = *(const uint4*)(hrow0 + kk + ak + 8);
      a[1].u[0] = *(const uint4*)(hrow1 + kk + ak);
      a[1].u[1] = *(const uint4*)(hrow1 + kk + ak + 8);

#pragma unroll
      for (int in = 0; in < 4; in++) {
        int ncol = wn * 64 + in * 16 + bn;       // 0..255 within chunk
        int t2 = ncol >> 7, rrow = ncol & 127;
        ABfrag bd;
        const __bf16* bdr = &WdR[coff + t2 * TILE + rrow * 32 + bk];
        bd.u[0] = *(const uint4*)(bdr);
        bd.u[1] = *(const uint4*)(bdr + 8);
#pragma unroll
        for (int im = 0; im < 2; im++)
          yacc[im][in] = wmma_bf16(a[im].v, bd.v, yacc[im][in]);
      }

      coff = noff;
      noff = (noff == 2 * SLOT) ? 0u : noff + SLOT;
    }

#pragma unroll
    for (int im = 0; im < 2; im++)
#pragma unroll
      for (int in = 0; in < 4; in++)
#pragma unroll
        for (int j = 0; j < 8; j++) {
          int m = wm * 32 + im * 16 + j + mhi;
          if (m < nrow) {
            int tk = toks[m];
            float wv = wrow[m];
            int n = wn * 64 + in * 16 + (lane & 15);
            unsafeAtomicAdd(&out[(size_t)tk * D + dbase + n], wv * yacc[im][in][j]);
          }
        }
  }
}

// ---------------------------------------------------------------------------
extern "C" void kernel_launch(void* const* d_in, const int* in_sizes, int n_in,
                              void* d_out, int out_size, void* d_ws, size_t ws_size,
                              hipStream_t stream) {
  const float* x      = (const float*)d_in[0];
  const float* rw     = (const float*)d_in[1];
  const float* gate_w = (const float*)d_in[2];
  const float* up_w   = (const float*)d_in[3];
  const float* down_w = (const float*)d_in[4];
  float* out = (float*)d_out;
  char* ws = (char*)d_ws;

  __bf16* xb     = (__bf16*)(ws + XB_OFF);
  __bf16* gate_b = (__bf16*)(ws + GATE_OFF);
  __bf16* up_b   = (__bf16*)(ws + UP_OFF);
  __bf16* down_b = (__bf16*)(ws + DOWN_OFF);
  float*  wfull  = (float*)(ws + WFULL_OFF);
  int*    counts = (int*)(ws + CNT_OFF);
  int*    offs   = (int*)(ws + OFFS_OFF);
  int*    toks   = (int*)(ws + TOK_OFF);
  __bf16* h_buf  = (__bf16*)(ws + HBUF_OFF);

  (void)hipMemsetAsync(d_out, 0, (size_t)out_size * sizeof(float), stream);
  (void)hipMemsetAsync(counts, 0, E * sizeof(int), stream);

  unsigned nx = (unsigned)(((size_t)T * D) / 1024);
  unsigned nw = (unsigned)(((size_t)E * F * D) / 1024);
  cvt_bf16_kernel<<<nx, 256, 0, stream>>>(x, xb, (size_t)T * D);
  cvt_bf16_kernel<<<nw, 256, 0, stream>>>(gate_w, gate_b, (size_t)E * F * D);
  cvt_bf16_kernel<<<nw, 256, 0, stream>>>(up_w, up_b, (size_t)E * F * D);
  cvt_bf16_kernel<<<nw, 256, 0, stream>>>(down_w, down_b, (size_t)E * F * D);

  float* logits_out = out + (size_t)T * D;
  router_kernel<<<T / 8, 256, 0, stream>>>(x, rw, logits_out, wfull, counts, toks);
  scan_kernel<<<1, 32, 0, stream>>>(counts, offs);

  dim3 gtile(T / 64, E);
  ffn1_kernel<<<gtile, 256, 0, stream>>>(xb, gate_b, up_b, toks, counts, offs, h_buf);
  ffn2_kernel<<<gtile, 256, 0, stream>>>(h_buf, down_b, toks, counts, offs, wfull, out);
}
